// TCTPLearner_58033598103767
// MI455X (gfx1250) — matrix operand order repaired and checked by reading, not compiled, not run on hardware
//
#include <hip/hip_runtime.h>
#include <hip/hip_bf16.h>
#include <math.h>

typedef __attribute__((ext_vector_type(16))) __bf16 v16bf;
typedef __attribute__((ext_vector_type(8)))  __bf16 v8bf;
typedef __attribute__((ext_vector_type(8)))  float  v8f;
typedef __attribute__((ext_vector_type(4)))  unsigned int v4u;
typedef __attribute__((ext_vector_type(8)))  int v8i;
typedef __attribute__((ext_vector_type(4)))  int v4i;

#define D_DIM 768
#define B_DIM 256
#define V_DIM 50257
#define U_DIM 1000
#define U_PAD 1008      /* 63 tiles of 16 */
#define U_TILES 63
#define Q_DIM 10000     /* 625 tiles of 16 exactly */
#define TOPK  8
#define INV_TEMP (1.0f / 0.07f)
#define EPS_NORM 1e-12f
#define NEG_BIG (-1e30f)

#define A_TILE_BYTES  (16 * D_DIM * 2)   /* 24576 */
#define B_TILE_BYTES  (16 * D_DIM * 2)   /* 24576 */
#define B_TILE_ELEMS  (16 * D_DIM)

#if __has_builtin(__builtin_amdgcn_tensor_load_to_lds)
#define HAVE_TDM 1
#else
#define HAVE_TDM 0
#endif

#if __has_builtin(__builtin_amdgcn_s_wait_tensorcnt)
#define TWAIT(n) __builtin_amdgcn_s_wait_tensorcnt(n)
#else
#define TWAIT(n) asm volatile("s_wait_tensorcnt " #n ::: "memory")
#endif

__device__ __forceinline__ __bf16 f2bf(float f) {
    unsigned u = __builtin_bit_cast(unsigned, f);
    unsigned r = (u + 0x7FFFu + ((u >> 16) & 1u)) >> 16;  // round-to-nearest-even
    unsigned short s = (unsigned short)r;
    return __builtin_bit_cast(__bf16, s);
}

#if HAVE_TDM
// Issue a TDM DMA of one 16x768 bf16 tile (row-major, stride 768) into LDS.
// D# layout per CDNA5 ISA ch.8: group0 = {count|flags, lds_addr, global_addr, type=2},
// group1 = {mask/data_size, dims/tile dims/strides}. Groups 2/3 zero (2D tensor).
__device__ __forceinline__ void tdm_load_tile(const __bf16* gsrc, unsigned lds_off) {
    unsigned long long ga = (unsigned long long)(const void*)gsrc;
    unsigned ga_lo = (unsigned)(ga & 0xFFFFFFFFu);
    unsigned ga_hi = (unsigned)((ga >> 32) & 0x01FFFFFFu) | (2u << 30); // type=2 ("image")
    v4u g0;
    g0[0] = 1u;                                             // count=1, user desc
    g0[1] = (unsigned)__builtin_amdgcn_readfirstlane((int)lds_off);
    g0[2] = (unsigned)__builtin_amdgcn_readfirstlane((int)ga_lo);
    g0[3] = (unsigned)__builtin_amdgcn_readfirstlane((int)ga_hi);
    v8i g1;
    g1[0] = (int)(1u << 16);      // workgroup_mask=0 (not in cluster), data_size=1 -> 2 bytes
    g1[1] = (int)(768u << 16);    // [63:48] tensor_dim0 lo16 = 768
    g1[2] = (int)(16u << 16);     // [79:64] tensor_dim0 hi16 = 0 ; [95:80] tensor_dim1 lo16 = 16
    g1[3] = (int)(768u << 16);    // [111:96] tensor_dim1 hi16 = 0 ; [127:112] tile_dim0 = 768
    g1[4] = 16;                   // [143:128] tile_dim1 = 16 ; tile_dim2 = 0
    g1[5] = 768;                  // [191:160] tensor_dim0_stride lo32 = 768
    g1[6] = 0;                    // stride hi16 / tensor_dim1_stride lo16
    g1[7] = 0;
    v4i gz = {0, 0, 0, 0};
#if defined(__clang_major__) && (__clang_major__ >= 23)
    v8i gz8 = {0, 0, 0, 0, 0, 0, 0, 0};
    __builtin_amdgcn_tensor_load_to_lds(g0, g1, gz, gz, gz8, 0);
#else
    __builtin_amdgcn_tensor_load_to_lds(g0, g1, gz, gz, 0);
#endif
}
#endif

// ---------------- per-row inverse norms: rn[row] = 1/max(||X_row||, eps) -----
__global__ void __launch_bounds__(256)
row_rnorm_kernel(const float* __restrict__ X, float* __restrict__ rn, int rows) {
    int lane = threadIdx.x & 31;
    int wave = threadIdx.x >> 5;
    int row  = blockIdx.x * 8 + wave;
    if (row >= rows) return;
    const float* p = X + (size_t)row * D_DIM;
    float s = 0.f;
    for (int i = lane; i < D_DIM; i += 32) { float v = p[i]; s += v * v; }
    #pragma unroll
    for (int off = 16; off > 0; off >>= 1) s += __shfl_xor(s, off, 32);
    if (lane == 0) rn[row] = 1.0f / fmaxf(sqrtf(s), EPS_NORM);
}

// ---------------- pack normalized rows to bf16 (zero-fill padded rows) -------
__global__ void __launch_bounds__(256)
pack_bf16_kernel(const float* __restrict__ X, const float* __restrict__ rn,
                 __bf16* __restrict__ outb, float* __restrict__ outf,
                 int rows, int padRows) {
    size_t idx   = (size_t)blockIdx.x * blockDim.x + threadIdx.x;
    size_t total = (size_t)padRows * D_DIM;
    if (idx >= total) return;
    int row = (int)(idx / D_DIM);
    float v = 0.f;
    if (row < rows) v = X[idx] * rn[row];
    outb[idx] = f2bf(v);
    if (outf && row < rows) outf[idx] = v;
}

// ---------------- fused GEMM1 + row-max -> loss_proto partial ----------------
// One wave per 16-row M tile of w_norm. A tile normalized->bf16 staged in LDS.
// B tiles (t_norm bf16) are DMA'd by the Tensor Data Mover into an LDS double
// buffer (TENSORcnt pipelined), so the VMEM pipe stays free during WMMA.
__global__ void __launch_bounds__(32)
proto_maxsim_kernel(const float* __restrict__ W, const float* __restrict__ rnw,
                    const __bf16* __restrict__ Tb, float* __restrict__ accum) {
    __shared__ __align__(16) char smem[A_TILE_BYTES + 2 * B_TILE_BYTES];
    __bf16* aT = (__bf16*)smem;
    __bf16* bB = (__bf16*)(smem + A_TILE_BYTES);

    const int lane  = threadIdx.x;
    const int mbase = blockIdx.x * 16;

    // stage normalized A tile (bf16) into LDS
    for (int idx = lane; idx < 16 * D_DIM; idx += 32) {
        int r    = idx / D_DIM;
        int cidx = idx - r * D_DIM;
        int row  = mbase + r;
        float v  = (row < V_DIM) ? W[(size_t)row * D_DIM + cidx] * rnw[row] : 0.f;
        aT[idx]  = f2bf(v);
    }
    __syncthreads();

    const int m  = lane & 15;
    const int k0 = (lane >> 4) * 16;   // A: K sub-block per half-wave
    const int kb = (lane >> 4) * 16;   // B: K sub-block per half-wave

    float best[8];
    #pragma unroll
    for (int r = 0; r < 8; ++r) best[r] = NEG_BIG;

#if HAVE_TDM
    const unsigned ldsB_off = (unsigned)(uintptr_t)(smem + A_TILE_BYTES);
    tdm_load_tile(Tb, ldsB_off);                       // prologue: tile 0 -> buf 0
#else
    for (int idx = lane; idx < B_TILE_ELEMS; idx += 32) bB[idx] = Tb[idx];
#endif

    for (int ut = 0; ut < U_TILES; ++ut) {
        const int buf = ut & 1;
        if (ut + 1 < U_TILES) {
            const __bf16* nxt = Tb + (size_t)(ut + 1) * B_TILE_ELEMS;
#if HAVE_TDM
            tdm_load_tile(nxt, ldsB_off + (unsigned)((buf ^ 1) * B_TILE_BYTES));
#else
            __builtin_prefetch(nxt, 0, 3);
            for (int idx = lane; idx < B_TILE_ELEMS; idx += 32)
                bB[(buf ^ 1) * B_TILE_ELEMS + idx] = nxt[idx];
#endif
        }
#if HAVE_TDM
        if (ut + 1 < U_TILES) { TWAIT(1); } else { TWAIT(0); }
#else
        __syncthreads();
#endif
        v8f c = {};
        const __bf16* arow = aT + m * D_DIM;
        const __bf16* brow = bB + buf * B_TILE_ELEMS + m * D_DIM + kb;
        #pragma unroll 4
        for (int k = 0; k < D_DIM / 32; ++k) {
            v8bf lo = *(const v8bf*)(arow + k * 32 + k0);
            v8bf hi = *(const v8bf*)(arow + k * 32 + k0 + 8);
            v16bf a = __builtin_shufflevector(lo, hi, 0,1,2,3,4,5,6,7,8,9,10,11,12,13,14,15);
            v16bf b = *(const v16bf*)(brow + k * 32);
            c = __builtin_amdgcn_wmma_f32_16x16x32_bf16(false, a, false, b, (short)0, c, false, false);
        }
        bool n_ok = (ut * 16 + m) < U_DIM;   // mask padded prototype columns
        #pragma unroll
        for (int r = 0; r < 8; ++r) {
            float cand = n_ok ? c[r] : NEG_BIG;
            best[r] = fmaxf(best[r], cand);
        }
#if !HAVE_TDM
        __syncthreads();
#endif
    }

    // reduce max across the 16 lanes that share each output row
    #pragma unroll
    for (int off = 8; off > 0; off >>= 1) {
        #pragma unroll
        for (int r = 0; r < 8; ++r)
            best[r] = fmaxf(best[r], __shfl_xor(best[r], off, 32));
    }
    if ((lane & 15) == 0) {
        int rbase = (lane == 0) ? 0 : 8;   // lane0 -> rows 0..7, lane16 -> rows 8..15
        float part = 0.f;
        #pragma unroll
        for (int r = 0; r < 8; ++r) {
            int row = mbase + rbase + r;
            if (row < V_DIM) part += 2.f - 2.f * best[r];
        }
        atomicAdd(accum, part);
    }
}

// ---------------- GEMM2: sim_bq = ts_norm x sup_norm^T (bf16 WMMA) -----------
__global__ void __launch_bounds__(32)
simbq_kernel(const __bf16* __restrict__ TSb, const __bf16* __restrict__ SUPb,
             float* __restrict__ simbq) {
    const int lane  = threadIdx.x;
    const int nbase = blockIdx.x * 16;
    const int mbase = blockIdx.y * 16;
    const int m  = lane & 15;
    const int k0 = (lane >> 4) * 16;
    const int kb = (lane >> 4) * 16;
    const __bf16* arow = TSb  + (size_t)(mbase + m) * D_DIM;
    const __bf16* brow = SUPb + (size_t)(nbase + m) * D_DIM + kb;
    v8f c = {};
    #pragma unroll 4
    for (int k = 0; k < D_DIM / 32; ++k) {
        v8bf lo = *(const v8bf*)(arow + k * 32 + k0);
        v8bf hi = *(const v8bf*)(arow + k * 32 + k0 + 8);
        v16bf a = __builtin_shufflevector(lo, hi, 0,1,2,3,4,5,6,7,8,9,10,11,12,13,14,15);
        v16bf b = *(const v16bf*)(brow + k * 32);
        c = __builtin_amdgcn_wmma_f32_16x16x32_bf16(false, a, false, b, (short)0, c, false, false);
    }
    const int n = nbase + m;
    #pragma unroll
    for (int r = 0; r < 8; ++r) {
        int row = mbase + ((lane < 16) ? r : 8 + r);
        simbq[(size_t)row * Q_DIM + n] = c[r];
    }
}

// ---------------- top-8 per row via 8 masked block-argmax passes -------------
__global__ void __launch_bounds__(256)
topk_kernel(const float* __restrict__ simbq, int* __restrict__ topk) {
    __shared__ float sv[256];
    __shared__ int   si[256];
    __shared__ int   chosen[TOPK];
    int b = blockIdx.x, t = threadIdx.x;
    const float* row = simbq + (size_t)b * Q_DIM;
    for (int it = 0; it < TOPK; ++it) {
        float bv = NEG_BIG; int bi = -1;
        for (int c = t; c < Q_DIM; c += 256) {
            bool skip = false;
            for (int j = 0; j < it; ++j) skip |= (chosen[j] == c);
            if (!skip) { float v = row[c]; if (v > bv) { bv = v; bi = c; } }
        }
        sv[t] = bv; si[t] = bi; __syncthreads();
        for (int s = 128; s > 0; s >>= 1) {
            if (t < s && sv[t + s] > sv[t]) { sv[t] = sv[t + s]; si[t] = si[t + s]; }
            __syncthreads();
        }
        if (t == 0) { chosen[it] = si[0]; topk[b * TOPK + it] = si[0]; }
        __syncthreads();
    }
}

// ------- gather nearest_tctps (output), positive = l2norm(mean), pos_sim -----
__global__ void __launch_bounds__(256)
gather_pos_kernel(const float* __restrict__ SUP, const float* __restrict__ rnsup,
                  const int* __restrict__ topk, const float* __restrict__ tsn,
                  float* __restrict__ outNearest, float* __restrict__ pos_sim) {
    __shared__ int   idx[TOPK];
    __shared__ float posb[D_DIM];
    __shared__ float red[256];
    int b = blockIdx.x, t = threadIdx.x;
    if (t < TOPK) idx[t] = topk[b * TOPK + t];
    __syncthreads();
    float ssq = 0.f;
    for (int d = t; d < D_DIM; d += 256) {
        float s = 0.f;
        #pragma unroll
        for (int k = 0; k < TOPK; ++k) {
            int q = idx[k];
            float v = SUP[(size_t)q * D_DIM + d] * rnsup[q];
            outNearest[((size_t)b * TOPK + k) * D_DIM + d] = v;
            s += v;
        }
        float pr = s * (1.0f / TOPK);
        posb[d] = pr;
        ssq += pr * pr;
    }
    red[t] = ssq; __syncthreads();
    for (int s = 128; s > 0; s >>= 1) { if (t < s) red[t] += red[t + s]; __syncthreads(); }
    float scale = 1.0f / fmaxf(sqrtf(red[0]), EPS_NORM);
    __syncthreads();
    float dot = 0.f;
    for (int d = t; d < D_DIM; d += 256) dot += posb[d] * tsn[(size_t)b * D_DIM + d];
    red[t] = dot; __syncthreads();
    for (int s = 128; s > 0; s >>= 1) { if (t < s) red[t] += red[t + s]; __syncthreads(); }
    if (t == 0) pos_sim[b] = red[0] * scale * INV_TEMP;
}

// ---------------- NNCL: per-row logsumexp over [pos, neg_sim row] ------------
__global__ void __launch_bounds__(256)
nncl_kernel(const float* __restrict__ tsn, const float* __restrict__ pos_sim,
            float* __restrict__ accum) {
    __shared__ float red[256];
    int b = blockIdx.x, c = threadIdx.x;
    const float* xb = tsn + (size_t)b * D_DIM;
    const float* xc = tsn + (size_t)c * D_DIM;
    float acc = 0.f;
    for (int d = 0; d < D_DIM; ++d) acc += xb[d] * xc[d];
    float logit = (b == c) ? NEG_BIG : acc * INV_TEMP;
    float ps = pos_sim[b];
    red[c] = logit; __syncthreads();
    for (int s = 128; s > 0; s >>= 1) { if (c < s) red[c] = fmaxf(red[c], red[c + s]); __syncthreads(); }
    float m = fmaxf(red[0], ps);
    __syncthreads();
    red[c] = __expf(logit - m); __syncthreads();
    for (int s = 128; s > 0; s >>= 1) { if (c < s) red[c] += red[c + s]; __syncthreads(); }
    if (c == 0) {
        float lse = m + __logf(red[0] + __expf(ps - m));
        atomicAdd(accum + 1, lse - ps);
    }
}

__global__ void zero_kernel(float* accum) {
    if (threadIdx.x < 2) accum[threadIdx.x] = 0.f;
}

__global__ void finalize_kernel(const float* __restrict__ accum, float* __restrict__ dout) {
    if (threadIdx.x == 0) {
        size_t npos = (size_t)B_DIM * TOPK * D_DIM;
        dout[npos]     = accum[0] / ((float)V_DIM * (float)D_DIM);  // loss_proto
        dout[npos + 1] = accum[1] / (float)B_DIM;                   // loss_nncl
    }
}

static inline size_t alignup256(size_t x) { return (x + 255) & ~(size_t)255; }

extern "C" void kernel_launch(void* const* d_in, const int* in_sizes, int n_in,
                              void* d_out, int out_size, void* d_ws, size_t ws_size,
                              hipStream_t stream) {
    (void)in_sizes; (void)n_in; (void)out_size; (void)ws_size;
    const float* ts   = (const float*)d_in[0];  // [256, 768]
    const float* wemb = (const float*)d_in[1];  // [50257, 768]
    const float* tct  = (const float*)d_in[2];  // [1000, 768]
    const float* sup  = (const float*)d_in[3];  // [10000, 768]
    float* dout = (float*)d_out;

    char* p = (char*)d_ws;
    auto take = [&](size_t bytes) -> char* { char* r = p; p += alignup256(bytes); return r; };
    float*  rnw    = (float*)take((size_t)V_DIM * 4);
    float*  rnt    = (float*)take((size_t)U_DIM * 4);
    float*  rnts   = (float*)take((size_t)B_DIM * 4);
    float*  rnsup  = (float*)take((size_t)Q_DIM * 4);
    __bf16* tb     = (__bf16*)take((size_t)U_PAD * D_DIM * 2);
    __bf16* tsb    = (__bf16*)take((size_t)B_DIM * D_DIM * 2);
    __bf16* supb   = (__bf16*)take((size_t)Q_DIM * D_DIM * 2);
    float*  tsn    = (float*)take((size_t)B_DIM * D_DIM * 4);
    float*  simbq  = (float*)take((size_t)B_DIM * Q_DIM * 4);
    int*    topk   = (int*)take((size_t)B_DIM * TOPK * 4);
    float*  possim = (float*)take((size_t)B_DIM * 4);
    float*  accum  = (float*)take(2 * 4);

    zero_kernel<<<1, 32, 0, stream>>>(accum);
    row_rnorm_kernel<<<(V_DIM + 7) / 8, 256, 0, stream>>>(wemb, rnw, V_DIM);
    row_rnorm_kernel<<<(U_DIM + 7) / 8, 256, 0, stream>>>(tct, rnt, U_DIM);
    row_rnorm_kernel<<<(B_DIM + 7) / 8, 256, 0, stream>>>(ts, rnts, B_DIM);
    row_rnorm_kernel<<<(Q_DIM + 7) / 8, 256, 0, stream>>>(sup, rnsup, Q_DIM);
    pack_bf16_kernel<<<(unsigned)(((size_t)U_PAD * D_DIM + 255) / 256), 256, 0, stream>>>(tct, rnt, tb, nullptr, U_DIM, U_PAD);
    pack_bf16_kernel<<<(unsigned)(((size_t)B_DIM * D_DIM + 255) / 256), 256, 0, stream>>>(ts, rnts, tsb, tsn, B_DIM, B_DIM);
    pack_bf16_kernel<<<(unsigned)(((size_t)Q_DIM * D_DIM + 255) / 256), 256, 0, stream>>>(sup, rnsup, supb, nullptr, Q_DIM, Q_DIM);
    proto_maxsim_kernel<<<(V_DIM + 15) / 16, 32, 0, stream>>>(wemb, rnw, tb, accum);
    simbq_kernel<<<dim3(Q_DIM / 16, B_DIM / 16), 32, 0, stream>>>(tsb, supb, simbq);
    topk_kernel<<<B_DIM, 256, 0, stream>>>(simbq, topk);
    gather_pos_kernel<<<B_DIM, 256, 0, stream>>>(sup, rnsup, topk, tsn, dout, possim);
    nncl_kernel<<<B_DIM, 256, 0, stream>>>(tsn, possim, accum);
    finalize_kernel<<<1, 32, 0, stream>>>(accum, dout);
}
